// MultiScaleModule_v2_27522150432908
// MI455X (gfx1250) — matrix operand order repaired and checked by reading, not compiled, not run on hardware
//
#include <hip/hip_runtime.h>
#include <math.h>

// ---------------------------------------------------------------------------
// MultiScale deformable-conv fusion for MI455X (gfx1250, wave32, WMMA).
// DCN einsum (M=256, K=2304, N=B*Ho*Wo) runs on v_wmma_f32_16x16x32_bf16
// with a bf16 hi/lo split (3 WMMAs per K=32 chunk) for ~fp32 accuracy.
// B tiles are staged through LDS with GLOBAL_LOAD_ASYNC_TO_LDS_B128
// (double-buffered, ASYNCcnt-tracked) when the toolchain exposes it.
// ---------------------------------------------------------------------------

#define BATCH 2
#define CH    256

typedef __attribute__((ext_vector_type(16))) __bf16 v16bf;
typedef __attribute__((ext_vector_type(8)))  float  v8f;
typedef __attribute__((ext_vector_type(4)))  int    v4i;

union FragU { uint4 q[2]; v16bf v; };
union FB32  { float f; unsigned u; };

#if __has_builtin(__builtin_amdgcn_global_load_async_to_lds_b128)
#define HAVE_ASYNC 1
#else
#define HAVE_ASYNC 0
#endif

#if HAVE_ASYNC
typedef __attribute__((address_space(1))) v4i* gv4i_p;   // global
typedef __attribute__((address_space(3))) v4i* lv4i_p;   // LDS
#endif

__device__ __forceinline__ void async_copy16(const void* g, void* l) {
#if HAVE_ASYNC
  __builtin_amdgcn_global_load_async_to_lds_b128((gv4i_p)g, (lv4i_p)l, 0, 0);
#else
  *(uint4*)l = *(const uint4*)g;
#endif
}

#if HAVE_ASYNC
#if __has_builtin(__builtin_amdgcn_s_wait_asynccnt)
#define WAIT_ASYNC(n) __builtin_amdgcn_s_wait_asynccnt(n)
#else
#define WAIT_ASYNC(n) asm volatile("s_wait_asynccnt %0" ::"i"(n))
#endif
#endif

__device__ __forceinline__ unsigned short f2bf(float f) {
  FB32 a; a.f = f;
  unsigned r = a.u + 0x7fffu + ((a.u >> 16) & 1u);   // RNE
  return (unsigned short)(r >> 16);
}
__device__ __forceinline__ float bf2f(unsigned short s) {
  FB32 a; a.u = ((unsigned)s) << 16; return a.f;
}

// ---------------------------------------------------------------------------
// 1) NCHW -> NHWC transpose, LDS tiled (32 x-positions x 32 channels)
// ---------------------------------------------------------------------------
__global__ void k_nchw2nhwc(const float* __restrict__ src, float* __restrict__ dst, int H) {
  __shared__ float tile[32][33];
  int nxt = (H + 31) >> 5;
  int bi = blockIdx.x;
  int ct = bi & 7; bi >>= 3;          // 8 channel tiles of 32
  int xt = bi % nxt; bi /= nxt;
  int y  = bi % H;  int b = bi / H;
  int tx = threadIdx.x, tc = threadIdx.y;
  for (int i = 0; i < 4; ++i) {
    int cl = tc * 4 + i;
    int xx = xt * 32 + tx;
    if (xx < H)
      tile[tx][cl] = src[((size_t)(b * CH + ct * 32 + cl) * H + y) * H + xx];
  }
  __syncthreads();
  for (int i = 0; i < 4; ++i) {
    int xl = tc * 4 + i;
    int xx = xt * 32 + xl;
    if (xx < H)
      dst[((size_t)(b * H + y) * H + xx) * CH + ct * 32 + tx] = tile[xl][tx];
  }
}

// ---------------------------------------------------------------------------
// 2) Bilinear resize NHWC -> NHWC (half-pixel centers, edge clamp)
// ---------------------------------------------------------------------------
__global__ void k_resize(const float* __restrict__ src, int Hin,
                         float* __restrict__ dst, int Hout) {
  int tid = blockIdx.x * blockDim.x + threadIdx.x;
  int total = BATCH * Hout * Hout * CH;
  if (tid >= total) return;
  int c = tid & 255; int rem = tid >> 8;
  int X = rem % Hout; rem /= Hout;
  int Y = rem % Hout; int b = rem / Hout;
  float scale = (float)Hin / (float)Hout;
  float sy = (Y + 0.5f) * scale - 0.5f;
  float sx = (X + 0.5f) * scale - 0.5f;
  float y0f = floorf(sy), x0f = floorf(sx);
  float wy = sy - y0f, wx = sx - x0f;
  int y0 = (int)y0f, x0 = (int)x0f;
  int y0c = min(max(y0, 0), Hin - 1), y1c = min(max(y0 + 1, 0), Hin - 1);
  int x0c = min(max(x0, 0), Hin - 1), x1c = min(max(x0 + 1, 0), Hin - 1);
  const float* sb = src + (size_t)b * Hin * Hin * CH + c;
  float v00 = sb[((size_t)y0c * Hin + x0c) * CH];
  float v01 = sb[((size_t)y0c * Hin + x1c) * CH];
  float v10 = sb[((size_t)y1c * Hin + x0c) * CH];
  float v11 = sb[((size_t)y1c * Hin + x1c) * CH];
  dst[tid] = (1.f - wy) * ((1.f - wx) * v00 + wx * v01)
           +        wy  * ((1.f - wx) * v10 + wx * v11);
}

// ---------------------------------------------------------------------------
// 3) offset(18) + sigmoid-mask(9) 3x3 conv, NHWC input. om: [B][27][H][W]
// ---------------------------------------------------------------------------
__global__ void k_offmask(const float* __restrict__ x, int H,
                          const float* __restrict__ ow, const float* __restrict__ ob,
                          const float* __restrict__ mw, const float* __restrict__ mb,
                          float* __restrict__ om) {
  int tid = blockIdx.x * blockDim.x + threadIdx.x;
  int total = BATCH * 27 * H * H;
  if (tid >= total) return;
  int oc = tid % 27; int rem = tid / 27;
  int xw = rem % H; rem /= H;
  int y  = rem % H; int b = rem / H;
  const float* wbase = (oc < 18) ? (ow + (size_t)oc * CH * 9)
                                 : (mw + (size_t)(oc - 18) * CH * 9);
  float s = (oc < 18) ? ob[oc] : mb[oc - 18];
  for (int kh = 0; kh < 3; ++kh) {
    int yy = y + kh - 1; if (yy < 0 || yy >= H) continue;
    for (int kw = 0; kw < 3; ++kw) {
      int xx = xw + kw - 1; if (xx < 0 || xx >= H) continue;
      const float* xp = x + ((size_t)(b * H + yy) * H + xx) * CH;
      const float* wp = wbase + kh * 3 + kw;
      float acc = 0.f;
      for (int ic = 0; ic < CH; ++ic) acc += xp[ic] * wp[(size_t)ic * 9];
      s += acc;
    }
  }
  if (oc >= 18) s = 1.f / (1.f + expf(-s));
  om[((size_t)(b * 27 + oc) * H + y) * H + xw] = s;
}

// ---------------------------------------------------------------------------
// 4) dcn weights -> bf16 hi/lo, K reordered as k = tap*256 + cin, row-major
// ---------------------------------------------------------------------------
__global__ void k_repack_w(const float* __restrict__ w,
                           unsigned short* __restrict__ hi,
                           unsigned short* __restrict__ lo) {
  int tid = blockIdx.x * blockDim.x + threadIdx.x;
  int total = 8 * CH * 2304;
  if (tid >= total) return;
  int kidx = tid % 2304;
  int m = (tid / 2304) % CH;
  int d = tid / (2304 * CH);
  int tap = kidx >> 8; int cin = kidx & 255;
  float v = w[(((size_t)(d * CH + m) * CH + cin) * 9) + tap];
  unsigned short h = f2bf(v);
  hi[tid] = h;
  lo[tid] = f2bf(v - bf2f(h));
}

// ---------------------------------------------------------------------------
// 5) Fused bilinear sample + mask modulate + bf16 split + B-fragment packing.
// Packed layout (linear == tid): ((cch*Ntiles + t)*32 + L)*16 + e, where
// lane L holds K = (e>>3)*16 + (L>=16)*8 + (e&7), N = L&15 of a 32x16 bf16 WMMA B tile.
// ---------------------------------------------------------------------------
__global__ void k_sample_pack(const float* __restrict__ x, int Hin,
                              const float* __restrict__ om, int Ho, int stride,
                              unsigned short* __restrict__ Shi,
                              unsigned short* __restrict__ Slo,
                              int Ntiles, int N) {
  int tid = blockIdx.x * blockDim.x + threadIdx.x;
  int total = 72 * Ntiles * 512;
  if (tid >= total) return;
  int e = tid & 15;
  int L = (tid >> 4) & 31;
  int g = tid >> 9;
  int t = g % Ntiles;
  int cch = g / Ntiles;
  int kk = ((e >> 3) << 4) + ((L >> 4) << 3) + (e & 7);
  int k = cch * 32 + kk;           // k = tap*256 + cin
  int tap = k >> 8;
  int c = k & 255;
  int n = t * 16 + (L & 15);
  float v = 0.f;
  if (n < N) {
    int HW = Ho * Ho;
    int b = n / HW; int rem = n - b * HW;
    int y = rem / Ho; int xx = rem - y * Ho;
    const float* omb = om + (size_t)b * 27 * HW;
    float offy = omb[(tap * 2 + 0) * HW + rem];
    float offx = omb[(tap * 2 + 1) * HW + rem];
    float msk  = omb[(18 + tap) * HW + rem];
    float py = (float)(y * stride + (tap / 3 - 1)) + offy;
    float px = (float)(xx * stride + (tap % 3 - 1)) + offx;
    float y0f = floorf(py), x0f = floorf(px);
    float wy = py - y0f, wx = px - x0f;
    int y0 = (int)y0f, x0 = (int)x0f;
    const float* xb = x + (size_t)b * Hin * Hin * CH + c;
    bool vy0 = (y0 >= 0) && (y0 < Hin);
    bool vy1 = (y0 + 1 >= 0) && (y0 + 1 < Hin);
    bool vx0 = (x0 >= 0) && (x0 < Hin);
    bool vx1 = (x0 + 1 >= 0) && (x0 + 1 < Hin);
    float s = 0.f;
    if (vy0 && vx0) s += (1.f - wy) * (1.f - wx) * xb[((size_t)y0 * Hin + x0) * CH];
    if (vy0 && vx1) s += (1.f - wy) * wx         * xb[((size_t)y0 * Hin + x0 + 1) * CH];
    if (vy1 && vx0) s += wy * (1.f - wx)         * xb[((size_t)(y0 + 1) * Hin + x0) * CH];
    if (vy1 && vx1) s += wy * wx                 * xb[((size_t)(y0 + 1) * Hin + x0 + 1) * CH];
    v = s * msk;
  }
  unsigned short h = f2bf(v);
  Shi[tid] = h;
  Slo[tid] = f2bf(v - bf2f(h));
}

// ---------------------------------------------------------------------------
// 6) WMMA GEMM: D[256 x N] += W[256 x 2304] * S[2304 x N] (+bias), bf16 split.
// Block tile 256x64: 8 waves x (32 rows x 64 cols). B chunk (32x64 hi+lo =
// 8KB) staged into LDS with async-to-LDS b128, double buffered; all waves
// read fragments from LDS. 24 WMMAs per wave per K-chunk, 72 chunks.
// ---------------------------------------------------------------------------
__global__ void __launch_bounds__(256) k_dcn_gemm(
    const unsigned short* __restrict__ Ahi, const unsigned short* __restrict__ Alo,
    const unsigned short* __restrict__ Shi, const unsigned short* __restrict__ Slo,
    const float* __restrict__ bias, float* __restrict__ out,
    int Ho, int Ntiles, int N) {
  __shared__ unsigned short sB[2][2][2048];   // [buf][hi/lo][4KB region]

  const int tid = threadIdx.x;
  const int lane = tid & 31;
  const int wave = tid >> 5;
  const int m0 = wave * 32;
  const int t0 = blockIdx.x * 4;             // first 16-col tile of this block

  const int l15 = lane & 15;
  const int hiHalf = lane >> 4;
  const size_t arow0 = (size_t)(m0 + l15) * 2304;
  const size_t arow1 = (size_t)(m0 + 16 + l15) * 2304;
  const int koBase = hiHalf * 8;

  const size_t chunkStride = (size_t)Ntiles * 512;     // elements per K-chunk
  const size_t gbase = (size_t)t0 * 512 + (size_t)tid * 8;

  v8f acc[2][4];
  v8f z = {};
#pragma unroll
  for (int mt = 0; mt < 2; ++mt)
#pragma unroll
    for (int j = 0; j < 4; ++j) acc[mt][j] = z;

  // preamble: stage chunk 0 into buffer 0 (each thread copies 16B of hi + lo)
  async_copy16(Shi + gbase, &sB[0][0][tid * 8]);
  async_copy16(Slo + gbase, &sB[0][1][tid * 8]);

  for (int cch = 0; cch < 72; ++cch) {
    const int buf = cch & 1;
    if (cch + 1 < 72) {
      size_t gnext = gbase + (size_t)(cch + 1) * chunkStride;
      async_copy16(Shi + gnext, &sB[buf ^ 1][0][tid * 8]);
      async_copy16(Slo + gnext, &sB[buf ^ 1][1][tid * 8]);
    }
#if HAVE_ASYNC
    if (cch + 1 < 72) { WAIT_ASYNC(2); } else { WAIT_ASYNC(0); }
#endif
    __syncthreads();

    // A fragments for this chunk (global; L2/WGP$-resident weights)
    const int ko = cch * 32 + koBase;
    FragU a0h, a0l, a1h, a1l;
    const uint4* p;
    p = (const uint4*)(Ahi + arow0 + ko); a0h.q[0] = p[0]; a0h.q[1] = p[2];
    p = (const uint4*)(Alo + arow0 + ko); a0l.q[0] = p[0]; a0l.q[1] = p[2];
    p = (const uint4*)(Ahi + arow1 + ko); a1h.q[0] = p[0]; a1h.q[1] = p[2];
    p = (const uint4*)(Alo + arow1 + ko); a1l.q[0] = p[0]; a1l.q[1] = p[2];

#pragma unroll
    for (int j = 0; j < 4; ++j) {
      FragU bh, bl;
      const uint4* q;
      q = (const uint4*)(&sB[buf][0][j * 512 + lane * 16]); bh.q[0] = q[0]; bh.q[1] = q[1];
      q = (const uint4*)(&sB[buf][1][j * 512 + lane * 16]); bl.q[0] = q[0]; bl.q[1] = q[1];

      acc[0][j] = __builtin_amdgcn_wmma_f32_16x16x32_bf16(false, a0h.v, false, bh.v, (short)0, acc[0][j], false, false);
      acc[0][j] = __builtin_amdgcn_wmma_f32_16x16x32_bf16(false, a0h.v, false, bl.v, (short)0, acc[0][j], false, false);
      acc[0][j] = __builtin_amdgcn_wmma_f32_16x16x32_bf16(false, a0l.v, false, bh.v, (short)0, acc[0][j], false, false);
      acc[1][j] = __builtin_amdgcn_wmma_f32_16x16x32_bf16(false, a1h.v, false, bh.v, (short)0, acc[1][j], false, false);
      acc[1][j] = __builtin_amdgcn_wmma_f32_16x16x32_bf16(false, a1h.v, false, bl.v, (short)0, acc[1][j], false, false);
      acc[1][j] = __builtin_amdgcn_wmma_f32_16x16x32_bf16(false, a1l.v, false, bh.v, (short)0, acc[1][j], false, false);
    }
    __syncthreads();   // protect buf reuse before next prefetch lands on it
  }

  // epilogue: accumulate into NCHW output region (+bias)
  const int HW = Ho * Ho;
#pragma unroll
  for (int j = 0; j < 4; ++j) {
    int n = (t0 + j) * 16 + l15;
    if (n < N) {
      int b = n / HW; int rem = n - b * HW;
      int y = rem / Ho; int x = rem - y * Ho;
#pragma unroll
      for (int mt = 0; mt < 2; ++mt) {
        int obase = m0 + mt * 16 + hiHalf * 8;
#pragma unroll
        for (int r = 0; r < 8; ++r) {
          int o = obase + r;
          size_t idx = ((size_t)(b * CH + o) * Ho + y) * Ho + x;
          out[idx] += acc[mt][j][r] + bias[o];
        }
      }
    }
  }
}

// ---------------------------------------------------------------------------
// Host orchestration
// ---------------------------------------------------------------------------
static inline int ceil_div(int a, int b) { return (a + b - 1) / b; }

extern "C" void kernel_launch(void* const* d_in, const int* in_sizes, int n_in,
                              void* d_out, int out_size, void* d_ws, size_t ws_size,
                              hipStream_t stream) {
  (void)in_sizes; (void)n_in; (void)out_size; (void)ws_size;
  const float* p_in[6];
  for (int i = 0; i < 6; ++i) p_in[i] = (const float*)d_in[i];
  const float* off_w  = (const float*)d_in[6];
  const float* off_b  = (const float*)d_in[7];
  const float* mask_w = (const float*)d_in[8];
  const float* mask_b = (const float*)d_in[9];
  const float* dcn_w  = (const float*)d_in[10];
  const float* dcn_b  = (const float*)d_in[11];
  float* out = (float*)d_out;

  const int S[6] = {160, 80, 40, 20, 10, 5};

  // workspace carve-out
  char* ws = (char*)d_ws;
  auto alloc = [&](size_t bytes) -> char* {
    char* p = ws;
    ws += (bytes + 255) & ~(size_t)255;
    return p;
  };
  float* nhwc[6];
  for (int i = 0; i < 6; ++i)
    nhwc[i] = (float*)alloc((size_t)BATCH * S[i] * S[i] * CH * 4);
  float* up[4];  // p4->80, p5->40, p6->20, p7->10
  const int upH[4] = {80, 40, 20, 10};
  for (int i = 0; i < 4; ++i)
    up[i] = (float*)alloc((size_t)BATCH * upH[i] * upH[i] * CH * 4);
  float* om[4];  // levels at 80,40,20,10
  const int lvlH[4] = {80, 40, 20, 10};
  for (int i = 0; i < 4; ++i)
    om[i] = (float*)alloc((size_t)BATCH * 27 * lvlH[i] * lvlH[i] * 4);
  unsigned short* Whi = (unsigned short*)alloc((size_t)8 * CH * 2304 * 2);
  unsigned short* Wlo = (unsigned short*)alloc((size_t)8 * CH * 2304 * 2);
  const size_t Smax = (size_t)2304 * (BATCH * 80 * 80);  // largest padded N (=12800)
  unsigned short* Shi = (unsigned short*)alloc(Smax * 2);
  unsigned short* Slo = (unsigned short*)alloc(Smax * 2);

  // 1) transposes
  for (int i = 0; i < 6; ++i) {
    int H = S[i], nxt = ceil_div(H, 32);
    dim3 grid(BATCH * H * nxt * 8), block(32, 8, 1);
    k_nchw2nhwc<<<grid, block, 0, stream>>>(p_in[i], nhwc[i], H);
  }
  // 2) resizes (sources p4..p7 -> one level up)
  for (int i = 0; i < 4; ++i) {
    int Hin = S[i + 2], Hout = upH[i];
    int total = BATCH * Hout * Hout * CH;
    k_resize<<<ceil_div(total, 256), 256, 0, stream>>>(nhwc[i + 2], Hin, up[i], Hout);
  }
  // 3) offsets + masks at p3..p6
  for (int l = 0; l < 4; ++l) {
    int H = lvlH[l];
    int total = BATCH * 27 * H * H;
    k_offmask<<<ceil_div(total, 256), 256, 0, stream>>>(
        nhwc[l + 1], H,
        off_w + (size_t)l * 18 * CH * 9, off_b + (size_t)l * 18,
        mask_w + (size_t)l * 9 * CH * 9, mask_b + (size_t)l * 9, om[l]);
  }
  // 4) weight repack
  {
    int total = 8 * CH * 2304;
    k_repack_w<<<ceil_div(total, 256), 256, 0, stream>>>(dcn_w, Whi, Wlo);
  }
  // 5) output init (p3..p6 identity terms, p7 pass-through)
  const size_t O3 = 0, O4 = 3276800, O5 = 4096000, O6 = 4300800, O7 = 4352000;
  (void)hipMemcpyAsync(out + O3, p_in[1], (size_t)BATCH * CH * 80 * 80 * 4, hipMemcpyDeviceToDevice, stream);
  (void)hipMemcpyAsync(out + O4, p_in[2], (size_t)BATCH * CH * 40 * 40 * 4, hipMemcpyDeviceToDevice, stream);
  (void)hipMemcpyAsync(out + O5, p_in[3], (size_t)BATCH * CH * 20 * 20 * 4, hipMemcpyDeviceToDevice, stream);
  (void)hipMemcpyAsync(out + O6, p_in[4], (size_t)BATCH * CH * 10 * 10 * 4, hipMemcpyDeviceToDevice, stream);
  (void)hipMemcpyAsync(out + O7, p_in[5], (size_t)BATCH * CH * 5 * 5 * 4, hipMemcpyDeviceToDevice, stream);

  // 6) eight DCN ops: {src, Hin, level, Ho, stride, dcn_idx, out_offset}
  struct Op { const float* src; int Hin; int lvl; int Ho; int stride; int d; size_t oofs; };
  const Op ops[8] = {
      {nhwc[0], 160, 0, 80, 2, 0, O3},  // p2_down
      {up[0],    80, 0, 80, 1, 2, O3},  // p4_up
      {nhwc[1],  80, 1, 40, 2, 1, O4},  // p3_down
      {up[1],    40, 1, 40, 1, 4, O4},  // p5_up
      {nhwc[2],  40, 2, 20, 2, 3, O5},  // p4_down
      {up[2],    20, 2, 20, 1, 6, O5},  // p6_up
      {nhwc[3],  20, 3, 10, 2, 5, O6},  // p5_down
      {up[3],    10, 3, 10, 1, 7, O6},  // p7_up
  };
  for (int i = 0; i < 8; ++i) {
    const Op& op = ops[i];
    int N = BATCH * op.Ho * op.Ho;
    int Nblk = ceil_div(N, 64);         // 256x64 GEMM blocks
    int Ntiles = Nblk * 4;              // 16-col tiles (padded to x4)
    int total = 72 * Ntiles * 512;
    k_sample_pack<<<ceil_div(total, 256), 256, 0, stream>>>(
        op.src, op.Hin, om[op.lvl], op.Ho, op.stride, Shi, Slo, Ntiles, N);
    k_dcn_gemm<<<Nblk, 256, 0, stream>>>(
        Whi + (size_t)op.d * CH * 2304, Wlo + (size_t)op.d * CH * 2304,
        Shi, Slo, dcn_b + (size_t)op.d * CH, out + op.oofs, op.Ho, Ntiles, N);
  }
}
